// LinearAttention_13125420056817
// MI455X (gfx1250) — matrix operand order repaired and checked by reading, not compiled, not run on hardware
//
#include <hip/hip_runtime.h>

// ---------------------------------------------------------------------------
// LinearAttention for MI455X (gfx1250, wave32, WMMA).
// GEMMs: bf16 inputs, f32 accumulate via v_wmma_f32_16x16x32_bf16.
// Staging: GLOBAL_LOAD_ASYNC_TO_LDS_B128 (ASYNCcnt) double-buffered pipeline,
// with a safe fallback to load+ds_store if the builtin is unavailable.
// LayerNorms / softmaxes computed in f32.
// ---------------------------------------------------------------------------

typedef __attribute__((ext_vector_type(16))) __bf16 v16bf;
typedef __attribute__((ext_vector_type(8)))  float  v8f;
typedef __attribute__((ext_vector_type(4)))  unsigned int u32x4;

union FragBF { v16bf v; u32x4 q[2]; };
union Pk8    { __bf16 h[8]; u32x4 q; };

#define BATCH 16
#define DIMC  384
#define NPIX  4096
#define QSCALE 0.102062072615966f   // (384/4)^-0.5

#if defined(__has_builtin)
#if __has_builtin(__builtin_amdgcn_global_load_async_to_lds_b128)
#define HAS_ASYNC_LDS 1
#endif
#endif

// 16B global -> LDS stage (async on CDNA5, sync fallback otherwise)
__device__ __forceinline__ void stage16(const __bf16* src, __bf16* dst) {
#ifdef HAS_ASYNC_LDS
    // builtin: (v4i addrspace(1)*, v4i addrspace(3)*, imm offset, imm cpol)
    typedef int v4i __attribute__((__vector_size__(4 * sizeof(int))));
    typedef __attribute__((address_space(1))) v4i g_v4i;
    typedef __attribute__((address_space(3))) v4i l_v4i;
    __builtin_amdgcn_global_load_async_to_lds_b128(
        (g_v4i*)src, (l_v4i*)dst, 0, 0);
#else
    *(u32x4*)dst = *(const u32x4*)src;
#endif
}

template<int N>
__device__ __forceinline__ void wait_async() {
#ifdef HAS_ASYNC_LDS
#if __has_builtin(__builtin_amdgcn_s_wait_asynccnt)
    __builtin_amdgcn_s_wait_asynccnt(N);
#else
    asm volatile("s_wait_asynccnt %0" :: "n"(N) : "memory");
#endif
#endif
}

// ---------------------------------------------------------------------------
// f32 -> bf16 convert (weights)
// ---------------------------------------------------------------------------
__global__ void cvt_f32_bf16(const float* __restrict__ src, __bf16* __restrict__ dst, int n) {
    int i = blockIdx.x * 256 + threadIdx.x;
    if (i < n) dst[i] = (__bf16)src[i];
}

// ---------------------------------------------------------------------------
// Channel LayerNorm (prenorm). Block = 32 pixels, 384-channel tile in LDS.
// Writes xn f32 [b][c][n] (coalesced) and xnT bf16 [b][n][c] (16B chunks).
// ---------------------------------------------------------------------------
__global__ void prenorm_kernel(const float* __restrict__ x, const float* __restrict__ g,
                               float* __restrict__ xn, __bf16* __restrict__ xnT) {
    __shared__ float tile[32][385];
    __shared__ float psum[8][32], psq[8][32];
    __shared__ float s_mean[32], s_rstd[32];
    int b  = blockIdx.x >> 7;
    int n0 = (blockIdx.x & 127) * 32;
    int tid = threadIdx.x;

    for (int i = tid; i < DIMC * 32; i += 256) {
        int c = i >> 5, p = i & 31;
        tile[p][c] = x[((size_t)(b * DIMC + c)) * NPIX + n0 + p];
    }
    __syncthreads();

    int p = tid & 31, cg = tid >> 5;
    float s = 0.f, sq = 0.f;
    for (int c = cg * 48; c < cg * 48 + 48; ++c) { float v = tile[p][c]; s += v; sq += v * v; }
    psum[cg][p] = s; psq[cg][p] = sq;
    __syncthreads();
    if (tid < 32) {
        float S = 0.f, Q = 0.f;
        for (int j = 0; j < 8; ++j) { S += psum[j][tid]; Q += psq[j][tid]; }
        float mean = S * (1.f / DIMC);
        float var  = Q * (1.f / DIMC) - mean * mean;
        s_mean[tid] = mean;
        s_rstd[tid] = rsqrtf(var + 1e-5f);
    }
    __syncthreads();

    for (int i = tid; i < DIMC * 32; i += 256) {
        int c = i >> 5, pp = i & 31;
        float v = (tile[pp][c] - s_mean[pp]) * s_rstd[pp] * g[c];
        tile[pp][c] = v;
        xn[((size_t)(b * DIMC + c)) * NPIX + n0 + pp] = v;
    }
    __syncthreads();

    for (int i = tid; i < 32 * 48; i += 256) {
        int pp = i / 48, sgm = i % 48;
        Pk8 pk;
#pragma unroll
        for (int j = 0; j < 8; ++j) pk.h[j] = (__bf16)tile[pp][sgm * 8 + j];
        *(u32x4*)&xnT[((size_t)(b * NPIX + n0 + pp)) * DIMC + sgm * 8] = pk.q;
    }
}

__device__ __forceinline__ void storeC(float*  p, float v) { *p = v; }
__device__ __forceinline__ void storeC(__bf16* p, float v) { *p = (__bf16)v; }

// ---------------------------------------------------------------------------
// Generic batched bf16 WMMA GEMM:  C[m][n] = sum_k A[m][k]*Bt[n][k] (+bias[m])
// A: bf16 [M][K] row-major.  Bt: bf16 [N][K] row-major (pre-transposed B).
// Tile 32(M) x BN(N); K-chunks of 32; double-buffered async LDS staging.
// Buffer selection is a runtime offset (+ #pragma unroll 1) so accumulator
// registers stay pinned (no cross-iteration v_mov copies).
// ---------------------------------------------------------------------------
template<int BN, typename CT>
__global__ void gemm_bf16(const __bf16* __restrict__ A, const __bf16* __restrict__ Bt,
                          CT* __restrict__ C, const float* __restrict__ bias,
                          int M, int N, int K, int ldc,
                          long aStride, long btStride,
                          long cStrideOuter, long cStrideInner, int cDiv) {
    constexpr int NW_N     = BN / 32;
    constexpr int NWAVES   = BN / 16;
    constexpr int NTHREADS = NWAVES * 32;
    constexpr int TOT      = (32 + BN) * 4;           // 16B segments per chunk
    constexpr int NITER    = (TOT + NTHREADS - 1) / NTHREADS;
    constexpr int ABUF     = 32 * 40;                 // elements per A buffer
    constexpr int BBUF     = BN * 40;                 // elements per B buffer

    __shared__ __align__(16) __bf16 Alds[2][32][40];
    __shared__ __align__(16) __bf16 Blds[2][BN][40];

    int tid  = threadIdx.x;
    int wave = tid >> 5, lane = tid & 31, lrow = lane & 15;
    int kA = (lane < 16) ? 0 : 16;
    int kB = (lane < 16) ? 0 : 16;

    int batch = blockIdx.z;
    const __bf16* Ab = A  + (size_t)batch * aStride;
    const __bf16* Bb = Bt + (size_t)batch * btStride;
    CT* Cb = C + (size_t)(batch / cDiv) * cStrideOuter + (size_t)(batch % cDiv) * cStrideInner;

    int mb = blockIdx.y * 32;
    int nb = blockIdx.x * BN;
    int mt = wave / NW_N;
    int np = wave % NW_N;

    // ---- precompute staging descriptors (K-invariant, buffer 0) ----
    const __bf16* sbase[NITER];
    __bf16* dbase[NITER];
    int dstride[NITER];                               // buffer-1 element offset
#pragma unroll
    for (int j = 0; j < NITER; ++j) {
        int i = tid + j * NTHREADS;
        if (i >= TOT) i -= TOT;                       // wrap: uniform issue count
        if (i < 32 * 4) {
            int r = i >> 2, sg = i & 3;
            sbase[j] = Ab + (size_t)(mb + r) * K + sg * 8;
            dbase[j] = &Alds[0][r][sg * 8];
            dstride[j] = ABUF;
        } else {
            int ii = i - 32 * 4;
            int r = ii >> 2, sg = ii & 3;
            sbase[j] = Bb + (size_t)(nb + r) * K + sg * 8;
            dbase[j] = &Blds[0][r][sg * 8];
            dstride[j] = BBUF;
        }
    }
    auto issue = [&](int buf, int kc) {
#pragma unroll
        for (int j = 0; j < NITER; ++j)
            stage16(sbase[j] + kc, dbase[j] + buf * dstride[j]);
    };

    // fragment base pointers (buffer 0)
    const __bf16* aFrag  = &Alds[0][mt * 16 + lrow][0];
    const __bf16* bFrag0 = &Blds[0][np * 32 + lrow][0];
    const __bf16* bFrag1 = bFrag0 + 16 * 40;

    v8f acc0 = {}, acc1 = {};
    const int NC = K >> 5;

    issue(0, 0);
#pragma unroll 1
    for (int c = 0; c < NC; ++c) {
        int cur = c & 1;
        if (c + 1 < NC) { issue(cur ^ 1, (c + 1) << 5); wait_async<NITER>(); }
        else            { wait_async<0>(); }
        __syncthreads();

        int ao = cur * ABUF, bo = cur * BBUF;
        FragBF a, b0, b1;
        a.q[0]  = *(const u32x4*)(aFrag  + ao + kA);
        a.q[1]  = *(const u32x4*)(aFrag  + ao + kA + 8);
        b0.q[0] = *(const u32x4*)(bFrag0 + bo + kB);
        b0.q[1] = *(const u32x4*)(bFrag0 + bo + kB + 8);
        b1.q[0] = *(const u32x4*)(bFrag1 + bo + kB);
        b1.q[1] = *(const u32x4*)(bFrag1 + bo + kB + 8);

        acc0 = __builtin_amdgcn_wmma_f32_16x16x32_bf16(false, a.v, false, b0.v,
                                                       (short)0, acc0, false, false);
        acc1 = __builtin_amdgcn_wmma_f32_16x16x32_bf16(false, a.v, false, b1.v,
                                                       (short)0, acc1, false, false);
        __syncthreads();
    }

    int rbase = mb + mt * 16 + ((lane < 16) ? 0 : 8);
    int c0 = nb + np * 32 + lrow;
#pragma unroll
    for (int v = 0; v < 8; ++v) {
        int row = rbase + v;
        float bv = bias ? bias[row] : 0.f;
        storeC(Cb + (size_t)row * ldc + c0,      acc0[v] + bv);
        storeC(Cb + (size_t)row * ldc + c0 + 16, acc1[v] + bv);
    }
}

// ---------------------------------------------------------------------------
// q softmax over the 96 per-head channels, per pixel -> qT bf16 [bh][n][96]
// ---------------------------------------------------------------------------
__global__ void q_softmax_kernel(const __bf16* __restrict__ qkv, __bf16* __restrict__ qT) {
    int bh = blockIdx.y;
    int b = bh >> 2, h = bh & 3;
    int n = blockIdx.x * 256 + threadIdx.x;
    const __bf16* base = qkv + ((size_t)(b * 1152 + h * 96)) * NPIX + n;
    float m = -1e30f;
    for (int d = 0; d < 96; ++d) m = fmaxf(m, (float)base[(size_t)d * NPIX]);
    float s = 0.f;
    for (int d = 0; d < 96; ++d) s += __expf((float)base[(size_t)d * NPIX] - m);
    float inv = QSCALE / s;
    __bf16* outp = qT + ((size_t)bh * NPIX + n) * 96;
    for (int d = 0; d < 96; ++d)
        outp[d] = (__bf16)(__expf((float)base[(size_t)d * NPIX] - m) * inv);
}

// ---------------------------------------------------------------------------
// k softmax over 4096 spatial positions per (b,h,d) row, in place in qkvBf.
// ---------------------------------------------------------------------------
__global__ void k_softmax_kernel(__bf16* __restrict__ qkv) {
    __shared__ float red[256];
    int r = blockIdx.x;
    int bh = r / 96, d = r % 96;
    int b = bh >> 2, h = bh & 3;
    __bf16* row = qkv + ((size_t)(b * 1152 + 384 + h * 96 + d)) * NPIX;
    int tid = threadIdx.x;
    float vals[16];
    float lm = -1e30f;
    for (int j = 0; j < 16; ++j) { vals[j] = (float)row[tid + j * 256]; lm = fmaxf(lm, vals[j]); }
    red[tid] = lm; __syncthreads();
    for (int s2 = 128; s2 > 0; s2 >>= 1) {
        if (tid < s2) red[tid] = fmaxf(red[tid], red[tid + s2]);
        __syncthreads();
    }
    float m = red[0]; __syncthreads();
    float ls = 0.f;
    for (int j = 0; j < 16; ++j) { vals[j] = __expf(vals[j] - m); ls += vals[j]; }
    red[tid] = ls; __syncthreads();
    for (int s2 = 128; s2 > 0; s2 >>= 1) {
        if (tid < s2) red[tid] += red[tid + s2];
        __syncthreads();
    }
    float inv = 1.f / red[0];
    for (int j = 0; j < 16; ++j) row[tid + j * 256] = (__bf16)(vals[j] * inv);
}

// ---------------------------------------------------------------------------
// context[d][e] = sum_n k[d][n]*v[e][n] over 4096 pixels, per (b,h).
// Waves 0..5 own 16-row d-strips x all 6 e-tiles. Async double-buffered
// staging; A frag + all six B frags loaded, then 6 back-to-back WMMAs.
// Runtime buffer offset + no-unroll keeps accumulators pinned.
// Writes context transposed: ctxT[bh][e][d].
// ---------------------------------------------------------------------------
__global__ void context_kernel(const __bf16* __restrict__ qkv, __bf16* __restrict__ ctxT) {
    constexpr int TOT   = 96 * 4 * 2;     // 768 16B segments per chunk
    constexpr int NITER = 3;              // 768 / 256
    constexpr int MBUF  = 96 * 40;        // elements per K/V buffer
    __shared__ __align__(16) __bf16 Klds[2][96][40];
    __shared__ __align__(16) __bf16 Vlds[2][96][40];
    int bh = blockIdx.x;
    int b = bh >> 2, h = bh & 3;
    const __bf16* kbase = qkv + ((size_t)(b * 1152 + 384 + h * 96)) * NPIX;
    const __bf16* vbase = qkv + ((size_t)(b * 1152 + 768 + h * 96)) * NPIX;
    int tid = threadIdx.x, wave = tid >> 5, lane = tid & 31, lrow = lane & 15;
    int kA = (lane < 16) ? 0 : 16;
    int kB = (lane < 16) ? 0 : 16;

    const __bf16* sbase[NITER];
    __bf16* dbase[NITER];
#pragma unroll
    for (int j = 0; j < NITER; ++j) {
        int i = tid + j * 256;
        if (i >= TOT) i -= TOT;
        int mat = (i >= 384);
        int ii = mat ? i - 384 : i;
        int r = ii >> 2, sg = ii & 3;
        sbase[j] = (mat ? vbase : kbase) + (size_t)r * NPIX + sg * 8;
        dbase[j] = mat ? &Vlds[0][r][sg * 8] : &Klds[0][r][sg * 8];
    }
    auto issue = [&](int buf, int nc) {
#pragma unroll
        for (int j = 0; j < NITER; ++j)
            stage16(sbase[j] + nc, dbase[j] + buf * MBUF);
    };

    const __bf16* aFrag = &Klds[0][wave * 16 + lrow][0];
    const __bf16* bFrag = &Vlds[0][lrow][0];

    v8f acc[6] = {};
    const int NC = NPIX >> 5;             // 128 chunks

    issue(0, 0);
#pragma unroll 1
    for (int c = 0; c < NC; ++c) {
        int cur = c & 1;
        if (c + 1 < NC) { issue(cur ^ 1, (c + 1) << 5); wait_async<NITER>(); }
        else            { wait_async<0>(); }
        __syncthreads();
        if (wave < 6) {
            int mo = cur * MBUF;
            FragBF a, bb[6];
            a.q[0] = *(const u32x4*)(aFrag + mo + kA);
            a.q[1] = *(const u32x4*)(aFrag + mo + kA + 8);
#pragma unroll
            for (int t = 0; t < 6; ++t) {
                bb[t].q[0] = *(const u32x4*)(bFrag + mo + t * (16 * 40) + kB);
                bb[t].q[1] = *(const u32x4*)(bFrag + mo + t * (16 * 40) + kB + 8);
            }
#pragma unroll
            for (int t = 0; t < 6; ++t)
                acc[t] = __builtin_amdgcn_wmma_f32_16x16x32_bf16(false, a.v, false, bb[t].v,
                                                                 (short)0, acc[t], false, false);
        }
        __syncthreads();
    }

    if (wave < 6) {
        __bf16* outp = ctxT + (size_t)bh * (96 * 96);
#pragma unroll
        for (int t = 0; t < 6; ++t) {
#pragma unroll
            for (int v = 0; v < 8; ++v) {
                int e = t * 16 + lrow;
                int d = wave * 16 + v + ((lane < 16) ? 0 : 8);
                outp[e * 96 + d] = (__bf16)acc[t][v];
            }
        }
    }
}

// ---------------------------------------------------------------------------
// Final: channel LN of (Wout@attn + bout), * outnorm_g, + xn residual.
// ---------------------------------------------------------------------------
__global__ void out_norm_kernel(const float* __restrict__ fin, const float* __restrict__ g,
                                const float* __restrict__ xn, float* __restrict__ out) {
    int b = blockIdx.y;
    int n = blockIdx.x * 256 + threadIdx.x;
    size_t base = (size_t)b * DIMC * NPIX + n;
    float s = 0.f, q = 0.f;
    for (int o = 0; o < DIMC; ++o) {
        float v = fin[base + (size_t)o * NPIX];
        s += v; q += v * v;
    }
    float mean = s * (1.f / DIMC);
    float var  = q * (1.f / DIMC) - mean * mean;
    float rstd = rsqrtf(var + 1e-5f);
    for (int o = 0; o < DIMC; ++o) {
        float v = fin[base + (size_t)o * NPIX];
        out[base + (size_t)o * NPIX] = (v - mean) * rstd * g[o] + xn[base + (size_t)o * NPIX];
    }
}

// ---------------------------------------------------------------------------
// Host-side pipeline. Workspace (bytes), with aliasing:
//   xn      f32  [16][384][4096]   @ 0          (100663296)
//   xnT     bf16 [16][4096][384]   @ 100663296  ( 50331648) -> reused as attnT
//   WqkvBf  bf16 [1152][384]       @ 150994944
//   WoutBf  bf16 [384][384]        @ 151879680
//   qkvBf   bf16 [16][1152][4096]  @ 152174592  (150994944) -> reused as fin(f32)
//   qT      bf16 [64][4096][96]    @ 303169536  ( 50331648)
//   ctxT    bf16 [64][96][96]      @ 353501184  (  1179648)
// ---------------------------------------------------------------------------
extern "C" void kernel_launch(void* const* d_in, const int* in_sizes, int n_in,
                              void* d_out, int out_size, void* d_ws, size_t ws_size,
                              hipStream_t stream) {
    (void)in_sizes; (void)n_in; (void)out_size; (void)ws_size;
    const float* x     = (const float*)d_in[0];
    const float* preg  = (const float*)d_in[1];
    const float* Wqkv  = (const float*)d_in[2];
    const float* Wout  = (const float*)d_in[3];
    const float* bout  = (const float*)d_in[4];
    const float* outg  = (const float*)d_in[5];
    float* out = (float*)d_out;

    char* ws = (char*)d_ws;
    float*  xn     = (float*)(ws + 0);
    __bf16* xnT    = (__bf16*)(ws + 100663296LL);
    __bf16* attnT  = (__bf16*)(ws + 100663296LL);   // aliases xnT (dead by then)
    __bf16* WqkvBf = (__bf16*)(ws + 150994944LL);
    __bf16* WoutBf = (__bf16*)(ws + 151879680LL);
    __bf16* qkvBf  = (__bf16*)(ws + 152174592LL);
    float*  fin    = (float*)(ws + 152174592LL);    // aliases qkvBf (dead by then)
    __bf16* qT     = (__bf16*)(ws + 303169536LL);
    __bf16* ctxT   = (__bf16*)(ws + 353501184LL);

    cvt_f32_bf16<<<(1152 * 384 + 255) / 256, 256, 0, stream>>>(Wqkv, WqkvBf, 1152 * 384);
    cvt_f32_bf16<<<(384 * 384 + 255) / 256, 256, 0, stream>>>(Wout, WoutBf, 384 * 384);

    prenorm_kernel<<<BATCH * 128, 256, 0, stream>>>(x, preg, xn, xnT);

    // QKV GEMM: qkvBf[b][o][n] = Wqkv @ xn   (M=1152, N=4096, K=384)
    gemm_bf16<128, __bf16><<<dim3(4096 / 128, 1152 / 32, BATCH), 256, 0, stream>>>(
        WqkvBf, xnT, qkvBf, nullptr, 1152, 4096, 384, 4096,
        0L, (long)4096 * 384, (long)1152 * 4096, 0L, 1);

    q_softmax_kernel<<<dim3(NPIX / 256, 64), 256, 0, stream>>>(qkvBf, qT);
    k_softmax_kernel<<<64 * 96, 256, 0, stream>>>(qkvBf);

    context_kernel<<<64, 256, 0, stream>>>(qkvBf, ctxT);

    // out GEMM: attnT[b][n][h*96+e] = sum_d qT[n][d]*ctxT[e][d]
    gemm_bf16<96, __bf16><<<dim3(1, 4096 / 32, 64), 192, 0, stream>>>(
        qT, ctxT, attnT, nullptr, 4096, 96, 96, 384,
        (long)4096 * 96, (long)96 * 96, (long)4096 * 384, 96L, 4);

    // Wout GEMM + bias: fin = Wout @ attn + bout  (M=384, N=4096, K=384)
    gemm_bf16<128, float><<<dim3(4096 / 128, 384 / 32, BATCH), 256, 0, stream>>>(
        WoutBf, attnT, fin, bout, 384, 4096, 384, 4096,
        0L, (long)4096 * 384, (long)384 * 4096, 0L, 1);

    out_norm_kernel<<<dim3(NPIX / 256, BATCH), 256, 0, stream>>>(fin, outg, xn, out);
}
